// DiffuseShader_82721070121226
// MI455X (gfx1250) — compile-verified, matrix operand order checked by reading, not run on hardware
//
#include <hip/hip_runtime.h>
#include <math.h>

typedef float    v2f  __attribute__((ext_vector_type(2)));
typedef float    v8f  __attribute__((ext_vector_type(8)));
typedef _Float16 v16h __attribute__((ext_vector_type(16)));

#ifndef __has_builtin
#define __has_builtin(x) 0
#endif

// ---------------------------------------------------------------------------
// K0: zero the per-point "empty surface" bitmask
// ---------------------------------------------------------------------------
__global__ void k_zero_mask(unsigned* __restrict__ em, int P) {
    int i = blockIdx.x * blockDim.x + threadIdx.x;
    if (i < P) em[i] = 0u;
}

// ---------------------------------------------------------------------------
// K1: per-point precompute.
//  - Rodrigues rotation R from v_loc=(0,-1,0) to u_hat=(l-p)/|l-p|
//  - d[p][s] = R @ hemi[s] + (l - p), stored as float4
//  - outputs col, opa, reflected_ray
//  - scatter empty bitmask: em[pointindex[p]%P] |= 1<<surf
// ---------------------------------------------------------------------------
__global__ void k_points(const float* __restrict__ p, const float* __restrict__ l,
                         const float* __restrict__ hemi,
                         const int* __restrict__ pointindex,
                         const int* __restrict__ indices,
                         const float* __restrict__ COL, const float* __restrict__ OPA,
                         float4* __restrict__ d4, unsigned* __restrict__ em,
                         float* __restrict__ outCol, float* __restrict__ outOpa,
                         float* __restrict__ outRray,
                         int P, int S, int M) {
    int i = blockIdx.x * blockDim.x + threadIdx.x;
    if (i >= P) return;

    float lx = l[0], ly = l[1], lz = l[2];
    float px = p[3*i], py = p[3*i+1], pz = p[3*i+2];
    float ux = lx - px, uy = ly - py, uz = lz - pz;
    float un = sqrtf(ux*ux + uy*uy + uz*uz);
    float hx = ux/un, hy = uy/un, hz = uz/un;     // u_hat

    float c  = -hy;                                // dot(v_loc, u_hat)
    float w0 = -hz, w1 = 0.0f, w2 = hx;            // cross(v_loc, u_hat)

    float vm[3][3] = { { 0.0f, -w2,   w1  },
                       { w2,    0.0f, -w0 },
                       { -w1,   w0,   0.0f } };
    float m2[3][3];
    #pragma unroll
    for (int a = 0; a < 3; ++a)
        #pragma unroll
        for (int b = 0; b < 3; ++b)
            m2[a][b] = vm[a][0]*vm[0][b] + vm[a][1]*vm[1][b] + vm[a][2]*vm[2][b];

    float opc = 1.0f + c;
    float R[3][3];
    #pragma unroll
    for (int a = 0; a < 3; ++a)
        #pragma unroll
        for (int b = 0; b < 3; ++b)
            R[a][b] = ((a == b) ? 1.0f : 0.0f) + vm[a][b] + m2[a][b] / opc;

    for (int s = 0; s < S; ++s) {
        float gx = hemi[3*s], gy = hemi[3*s+1], gz = hemi[3*s+2];
        float dx = R[0][0]*gx + R[0][1]*gy + R[0][2]*gz + ux;
        float dy = R[1][0]*gx + R[1][1]*gy + R[1][2]*gz + uy;
        float dz = R[2][0]*gx + R[2][1]*gy + R[2][2]*gz + uz;
        d4[(size_t)i*S + s] = make_float4(dx, dy, dz, 0.0f);
    }

    int pi   = pointindex[i];
    int surf = indices[2*pi + 0];
    int mat  = indices[2*pi + 1];
    size_t cm = ((size_t)surf * M + mat);
    outCol[3*i+0] = COL[cm*3+0];
    outCol[3*i+1] = COL[cm*3+1];
    outCol[3*i+2] = COL[cm*3+2];
    float op = OPA[cm];
    outOpa[i] = fminf(fmaxf(op, 0.0f), 1.0f);
    outRray[3*i+0] = ux; outRray[3*i+1] = uy; outRray[3*i+2] = uz;

    int local = pi % P;
    if (surf >= 0 && surf < 32) atomicOr(&em[local], 1u << surf);
}

// ---------------------------------------------------------------------------
// K2: per-triangle precompute: unit normal + plane offset k, and the
// barycentric constants B, D, E, G6=a1*c0-a0*c1, denG = D*(E*D + B*G6).
// ---------------------------------------------------------------------------
__global__ void k_tris(const float* __restrict__ V,
                       float4* __restrict__ nrm4, float4* __restrict__ triC, int N) {
    int n = blockIdx.x * blockDim.x + threadIdx.x;
    if (n >= N) return;
    const float* vv = V + (size_t)n * 12;
    float a0 = vv[0], a1 = vv[1], a2 = vv[2];
    float b0 = vv[3], b1 = vv[4], b2 = vv[5];
    float c0 = vv[6], c1 = vv[7], c2 = vv[8];
    float q0 = vv[9], q1 = vv[10], q2 = vv[11];

    float e1x = b0-a0, e1y = b1-a1, e1z = b2-a2;
    float e2x = c0-a0, e2y = c1-a1, e2z = c2-a2;
    float nx = e1y*e2z - e1z*e2y;
    float ny = e1z*e2x - e1x*e2z;
    float nz = e1x*e2y - e1y*e2x;
    float nl = sqrtf(nx*nx + ny*ny + nz*nz);
    nx /= nl; ny /= nl; nz /= nl;
    float kk = -(nx*q0 + ny*q1 + nz*q2);
    nrm4[n] = make_float4(nx, ny, nz, kk);

    float Bc = a0*b2 - a2*b0;
    float Dc = a0*b1 - a1*b0;
    float Ec = a0*c2 - a2*c0;
    float G6 = a1*c0 - a0*c1;
    float Fc = Bc*G6;
    float denG = Dc*(Ec*Dc + Fc);
    triC[3*n+0] = make_float4(a0, a1, a2, b0);
    triC[3*n+1] = make_float4(c0, Bc, Dc, G6);
    triC[3*n+2] = make_float4(denG, 0.0f, 0.0f, 0.0f);
}

// ---------------------------------------------------------------------------
// K3: VOK[p][j] = dot(p_pos[p], nrm[j]) + k[j] via WMMA. A = points [16x4]
// (K=3 padded), B = normals^T [4x16], C = k broadcast along rows.
// One wave per 16x16 tile; 8 waves / block.
// ---------------------------------------------------------------------------
__global__ void __launch_bounds__(256)
k_vok_wmma(const float* __restrict__ p, const float4* __restrict__ nrm4,
           float* __restrict__ VOK, int P, int N) {
    int wave = threadIdx.x >> 5;
    int lane = threadIdx.x & 31;
    int tilesN = N >> 4;
    int numTiles = (P >> 4) * tilesN;
    int tile = blockIdx.x * (blockDim.x >> 5) + wave;
    if (tile >= numTiles) return;          // wave-uniform exit, EXEC stays full

    int pt = tile / tilesN, nt = tile % tilesN;
    int row  = lane & 15;
    int hi   = lane >> 4;
    int pr   = (pt << 4) + row;            // A-matrix row (point index)
    int ncol = (nt << 4) + row;            // B/C/D column (triangle index)

    float  px = p[3*pr], py = p[3*pr+1], pz = p[3*pr+2];
    float4 nv = nrm4[ncol];

    v8f c;
    #pragma unroll
    for (int r = 0; r < 8; ++r) c[r] = nv.w;   // C[m][j] = k[j]

    v8f dacc;
#if __has_builtin(__builtin_amdgcn_wmma_f32_16x16x4_f32)
    // f32 A 16x4: lanes 0-15 hold K0,K1; lanes 16-31 hold K2,K3 (K3 = 0 pad)
    v2f a, b;
    a.x = hi ? pz   : px;   a.y = hi ? 0.0f : py;
    b.x = hi ? nv.z : nv.x; b.y = hi ? 0.0f : nv.y;
    dacc = __builtin_amdgcn_wmma_f32_16x16x4_f32(
        false, a, false, b, (short)0, c, false, false);
#elif __has_builtin(__builtin_amdgcn_wmma_f32_16x16x32_f16)
    // f16 fallback: K=32 with only K0..K2 nonzero (lanes 0-15 carry K0..7)
    v16h a, b;
    #pragma unroll
    for (int t = 0; t < 16; ++t) { a[t] = (_Float16)0; b[t] = (_Float16)0; }
    if (!hi) {
        a[0] = (_Float16)px;   a[1] = (_Float16)py;   a[2] = (_Float16)pz;
        b[0] = (_Float16)nv.x; b[1] = (_Float16)nv.y; b[2] = (_Float16)nv.z;
    }
    dacc = __builtin_amdgcn_wmma_f32_16x16x32_f16(
        false, a, false, b, (short)0, c, false, false);
#else
    #pragma unroll
    for (int r = 0; r < 8; ++r) {
        int m = r + hi*8;
        float qx = p[3*((pt<<4)+m)], qy = p[3*((pt<<4)+m)+1], qz = p[3*((pt<<4)+m)+2];
        dacc[r] = qx*nv.x + qy*nv.y + qz*nv.z + nv.w;
    }
    (void)c;
#endif
    // D layout: VGPR r -> M = r + 8*hi, N = lane&15
    int prBase = (pt << 4) + hi*8;
    #pragma unroll
    for (int r = 0; r < 8; ++r)
        VOK[(size_t)(prBase + r) * N + ncol] = dacc[r];
}

// scalar fallback if P or N not multiples of 16 (not the case for this harness)
__global__ void k_vok_scalar(const float* __restrict__ p, const float4* __restrict__ nrm4,
                             float* __restrict__ VOK, int P, int N) {
    size_t i = (size_t)blockIdx.x * blockDim.x + threadIdx.x;
    if (i >= (size_t)P * N) return;
    int pp = (int)(i / N), j = (int)(i % N);
    float4 nv = nrm4[j];
    VOK[i] = p[3*pp]*nv.x + p[3*pp+1]*nv.y + p[3*pp+2]*nv.z + nv.w;
}

// ---------------------------------------------------------------------------
// K4: main mask kernel. Block = (point p, flat chunk m in [ssk*N,(ssk+1)*N)).
// Then the torch-.view scramble gives ss = ssk (constant) and nn = e directly.
// 4 coalesced writes per thread, ray dirs staged in LDS.
// ---------------------------------------------------------------------------
__global__ void __launch_bounds__(256)
k_mask(const float* __restrict__ p, const float4* __restrict__ d4,
       const float4* __restrict__ nrm4, const float4* __restrict__ triC,
       const float* __restrict__ VOK, const unsigned* __restrict__ em,
       float* __restrict__ outMask, int P, int N, int S) {
    int blk = blockIdx.x;
    int pp  = blk / S;
    int ssk = blk % S;

    __shared__ float4 shd[64];                 // S <= 64 (S = 25 here)
    if ((int)threadIdx.x < S) shd[threadIdx.x] = d4[(size_t)pp*S + threadIdx.x];
    __syncthreads();

    float4 dk = shd[ssk];                      // denominator direction (block const)
    float ox = p[3*pp], oy = p[3*pp+1], oz = p[3*pp+2];
    unsigned emask = em[pp];
    const float* vokRow = VOK + (size_t)pp * N;
    float* outRow = outMask + (size_t)pp * N * S + (size_t)ssk * N;

    for (int e = threadIdx.x; e < N; e += blockDim.x) {
        float4 nv = nrm4[e];
        float denom = dk.x*nv.x + dk.y*nv.y + dk.z*nv.z;
        float t = -vokRow[e] / denom;

        int m = ssk * N + e;
        int n = (int)((unsigned)m / (unsigned)S);
        int s = m - n * S;

        bool emptyb = (n < 32) && ((emask >> n) & 1u);
        bool viable = (t > -1e-4f) && (t < 1.0f) && !emptyb;

        float4 ds = shd[s];
        float r0 = ox + t*ds.x, r1 = oy + t*ds.y, r2 = oz + t*ds.z;

        float4 t0 = triC[3*n+0], t1 = triC[3*n+1], t2 = triC[3*n+2];
        float a0 = t0.x, a1 = t0.y, a2 = t0.z, b0 = t0.w;
        float c0 = t1.x, Bc = t1.y, Dc = t1.z, G6 = t1.w;
        float denG = t2.x;

        float A  = a0*r2 - a2*r0;
        float C  = a0*r1 - a1*r0;
        float gam   = (Dc*(A*Dc - Bc*C)) / denG;
        float beta  = (C + gam*G6) / Dc;
        float alpha = (r0 - (beta*b0 + gam*c0)) / a0;

        float mv = (viable && (beta > 0.0f) && (gam > 0.0f) && (alpha > 0.0f))
                   ? 1.0f : 0.0f;
        outRow[e] = mv;
    }
}

// ---------------------------------------------------------------------------
// host launcher
// ---------------------------------------------------------------------------
extern "C" void kernel_launch(void* const* d_in, const int* in_sizes, int n_in,
                              void* d_out, int out_size, void* d_ws, size_t ws_size,
                              hipStream_t stream) {
    const float* V          = (const float*)d_in[0];
    const int*   indices    = (const int*)  d_in[1];
    const int*   pointindex = (const int*)  d_in[2];
    const float* COL        = (const float*)d_in[3];
    const float* OPA        = (const float*)d_in[4];
    const float* p          = (const float*)d_in[5];
    const float* l          = (const float*)d_in[6];
    // d_in[7] = normals (unused by reference), d_in[8] = it (unused)
    const float* hemi       = (const float*)d_in[9];

    int N = in_sizes[0] / 12;       // (N,4,3)
    int P = in_sizes[5] / 3;        // (P,3)
    int S = in_sizes[9] / 3;        // (S,3)
    int M = in_sizes[3] / (N * 3);  // (N,M,3)

    // workspace layout (float4 tables first for 16B alignment)
    float4*   d4   = (float4*)d_ws;              // P*S
    float4*   nrm4 = d4 + (size_t)P * S;         // N
    float4*   triC = nrm4 + N;                   // 3*N
    float*    VOK  = (float*)(triC + (size_t)3 * N);   // P*N
    unsigned* em   = (unsigned*)(VOK + (size_t)P * N); // P

    float* outMask = (float*)d_out;
    float* outCol  = outMask + (size_t)P * N * S;
    float* outOpa  = outCol + (size_t)3 * P;
    float* outRr   = outOpa + P;

    k_zero_mask<<<(P + 255) / 256, 256, 0, stream>>>(em, P);
    k_points<<<(P + 127) / 128, 128, 0, stream>>>(p, l, hemi, pointindex, indices,
                                                  COL, OPA, d4, em,
                                                  outCol, outOpa, outRr, P, S, M);
    k_tris<<<(N + 127) / 128, 128, 0, stream>>>(V, nrm4, triC, N);

    if ((P % 16 == 0) && (N % 16 == 0)) {
        int tiles  = (P / 16) * (N / 16);
        int blocks = (tiles + 7) / 8;            // 8 waves (tiles) per block
        k_vok_wmma<<<blocks, 256, 0, stream>>>(p, nrm4, VOK, P, N);
    } else {
        size_t tot = (size_t)P * N;
        k_vok_scalar<<<(unsigned)((tot + 255) / 256), 256, 0, stream>>>(p, nrm4, VOK, P, N);
    }

    k_mask<<<P * S, 256, 0, stream>>>(p, d4, nrm4, triC, VOK, em, outMask, P, N, S);
}
